// LSTM_TPA_attention_timeseries_13477607375058
// MI455X (gfx1250) — matrix operand order repaired and checked by reading, not compiled, not run on hardware
//
#include <hip/hip_runtime.h>
#include <hip/hip_bf16.h>
#include <math.h>

// ---------------------------------------------------------------------------
// LSTM + TPA attention + MLP head for MI455X (gfx1250, wave32, WMMA).
//
// Round-2 change: round-1 asm showed the register-resident LSTM weight slab
// (192 VGPR/lane) spilled to scratch (allocator gives ~128 VGPR/wave at 8
// waves/SIMD => gfx1250 RF ~1024 regs/lane/SIMD; 768KB can't be RF-resident).
// New scheme: weights are LDS-cached for waves 0-7 (192KB of the slab) and
// L2-streamed for waves 8-31 (the slab is shared by all 64 WGs and lives in
// the 192MB L2). Live registers per wave drop to ~70 -> no spills.
//
//  * k_lstm: 64 WGs x 1024 thr; fused G = [x_t | h] @ [W_ih|W_hh]^T per step
//    with v_wmma_f32_16x16x32_f16; h/c state in LDS, h re-packed as f16 A.
//  * k_tpa: one WG per batch row; hs[b] staged in LDS; hc / hc@W_tpa^T via
//    WMMA; alpha dot-reduction via ds_add_f32 + s_wait_dscnt.
//  * k_head: 16x16-tile WMMA GEMMs with fused bias/ReLU/BatchNorm epilogue.
// Workspace use: ~107 MB.
// ---------------------------------------------------------------------------

typedef __attribute__((ext_vector_type(16))) _Float16 v16h;
typedef __attribute__((ext_vector_type(8)))  float    v8f;

#define WMMA_F16(a, b, c)                                                    \
  __builtin_amdgcn_wmma_f32_16x16x32_f16(false, (a), false, (b), (short)0,   \
                                         (c), false, false)

#define B_   1024
#define T_   128
#define H_   256
#define G4_  1024   // 4*H (gate width, PyTorch order i,f,g,o)
#define DIN_ 128    // LSTM input dim (32+32 emb + 64 cont)
#define KC_  384    // DIN_ + H_ : fused K for [x_t | h]
#define FIN_ 672    // 160 nonseq + 256 seq_inp + 256 htprime
#define CW_  8      // waves whose weight slice is LDS-cached (8*24KB = 192KB)

// ---- CDNA5 WMMA f16 fragment index maps (ISA 7.12.2) ----------------------
// A 16x32 f16: lanes 0-15 row M=lane, K={0..7,16..23}; lanes 16-31 K={8..15,24..31}
static __device__ __forceinline__ int ak_idx(int lane, int h) {
  return ((lane & 16) ? 8 : 0) + (h & 7) + ((h & 8) << 1);
}
// B 32x16 f16: lanes 0-15 col N=lane, K={0..7,16..23}; lanes 16-31 K={8..15,24..31}
static __device__ __forceinline__ int bk_idx(int lane, int h) {
  return ((lane & 16) ? 8 : 0) + (h & 7) + ((h & 8) << 1);
}

// A fragment from row-major [M x K] block, leading dim ld (halves)
static __device__ __forceinline__ v16h load_a_rm(const _Float16* a, int ld,
                                                 int lane) {
  v16h r; int m = lane & 15;
  #pragma unroll
  for (int h = 0; h < 16; ++h) r[h] = a[m * ld + ak_idx(lane, h)];
  return r;
}
// B fragment from a weight stored row-major [N x K] (B[k][n] = w[n][k])
static __device__ __forceinline__ v16h load_b_wt(const _Float16* w, int ld,
                                                 int lane) {
  v16h r; int n = lane & 15;
  #pragma unroll
  for (int h = 0; h < 16; ++h) r[h] = w[n * ld + bk_idx(lane, h)];
  return r;
}
// C/D 16x16 f32: VGPR v -> row v (lanes 0-15) / v+8 (lanes 16-31), col lane&15
static __device__ __forceinline__ void store_c(float* d, int ld, int lane,
                                               v8f c) {
  int n = lane & 15, mb = (lane & 16) ? 8 : 0;
  #pragma unroll
  for (int v = 0; v < 8; ++v) d[(mb + v) * ld + n] = c[v];
}

static __device__ __forceinline__ float sigf(float x) {
  return 1.0f / (1.0f + __expf(-x));
}

// ---------------------------------------------------------------------------
// Setup kernels
// ---------------------------------------------------------------------------
__global__ void k_f32_to_f16(const float* __restrict__ src,
                             _Float16* __restrict__ dst, int n) {
  for (int i = blockIdx.x * blockDim.x + threadIdx.x; i < n;
       i += gridDim.x * blockDim.x)
    dst[i] = (_Float16)src[i];
}

// wcat[n][k] = (k<128 ? W_ih[n][k] : W_hh[n][k-128]); bsum = b_ih + b_hh
__global__ void k_build_wcat(const float* __restrict__ W_ih,
                             const float* __restrict__ W_hh,
                             const float* __restrict__ b_ih,
                             const float* __restrict__ b_hh,
                             _Float16* __restrict__ wcat,
                             float* __restrict__ bsum) {
  const int stride = gridDim.x * blockDim.x;
  for (int i = blockIdx.x * blockDim.x + threadIdx.x; i < G4_ * KC_;
       i += stride) {
    int n = i / KC_, k = i % KC_;
    float v = (k < DIN_) ? W_ih[n * DIN_ + k] : W_hh[n * H_ + (k - DIN_)];
    wcat[i] = (_Float16)v;
  }
  for (int i = blockIdx.x * blockDim.x + threadIdx.x; i < G4_; i += stride)
    bsum[i] = b_ih[i] + b_hh[i];
}

// seqF16[b,t,0:32]=emb_s0[cat0], [32:64]=emb_s1[cat1], [64:128]=seq_cont
__global__ void k_embed_seq(const float* __restrict__ seq_cont,
                            const int* __restrict__ seq_cat,
                            const float* __restrict__ emb_s0,
                            const float* __restrict__ emb_s1,
                            _Float16* __restrict__ seqF16) {
  const int tot = B_ * T_ * DIN_;
  for (int i = blockIdx.x * blockDim.x + threadIdx.x; i < tot;
       i += gridDim.x * blockDim.x) {
    int k = i & 127, bt = i >> 7;
    float v;
    if (k < 32)      v = emb_s0[seq_cat[bt * 2 + 0] * 32 + k];
    else if (k < 64) v = emb_s1[seq_cat[bt * 2 + 1] * 32 + (k - 32)];
    else             v = seq_cont[bt * 64 + (k - 64)];
    seqF16[i] = (_Float16)v;
  }
}

// fin[b,0:64]=emb_ns0, [64:128]=emb_ns1, [128:160]=ns_cont (f32->d_out, f16->ws)
__global__ void k_embed_nonseq(const int* __restrict__ ns_cat,
                               const float* __restrict__ ns_cont,
                               const float* __restrict__ emb_ns0,
                               const float* __restrict__ emb_ns1,
                               float* __restrict__ finOut,
                               _Float16* __restrict__ finF16) {
  const int tot = B_ * 160;
  for (int i = blockIdx.x * blockDim.x + threadIdx.x; i < tot;
       i += gridDim.x * blockDim.x) {
    int b = i / 160, k = i - b * 160;
    float v;
    if (k < 64)       v = emb_ns0[ns_cat[b * 2 + 0] * 64 + k];
    else if (k < 128) v = emb_ns1[ns_cat[b * 2 + 1] * 64 + (k - 64)];
    else              v = ns_cont[b * 32 + (k - 128)];
    finOut[(size_t)b * FIN_ + k] = v;
    finF16[(size_t)b * FIN_ + k] = (_Float16)v;
  }
}

// ---------------------------------------------------------------------------
// LSTM: 64 WGs x 1024 thr (32 waves). Wave w owns gate columns 32w..32w+31.
// Waves < CW_ read their weight slice from LDS; the rest stream from L2.
// ---------------------------------------------------------------------------
__global__ void __launch_bounds__(1024)
k_lstm(const _Float16* __restrict__ seqF16, const _Float16* __restrict__ wcat,
       const float* __restrict__ bsum, _Float16* __restrict__ hsF16,
       float* __restrict__ hn) {
  extern __shared__ char smem[];
  _Float16* sA = (_Float16*)smem;                              // [16][384]  12KB
  float*    sG = (float*)(smem + 16 * KC_ * 2);                // [16][1024] 64KB
  float*    sC = (float*)(smem + 16 * KC_ * 2 + 16 * G4_ * 4); // [16][256]  16KB
  _Float16* sW = (_Float16*)(smem + 16 * KC_ * 2 + 16 * G4_ * 4 + 16 * H_ * 4);
  // sW: first CW_*32 weight rows, row-major K (CW_*32*384 halves = 192KB)

  const int tid = threadIdx.x, lane = tid & 31, w = tid >> 5;
  const int b0 = blockIdx.x * 16;
  const bool cached = (w < CW_);

  // stage the LDS-cached prefix of the weight slab (rows 0..CW_*32)
  for (int i = tid; i < CW_ * 32 * KC_; i += 1024) sW[i] = wcat[i];

  // h0 = c0 = 0 (h lives inside the A staging buffer at K=128..383)
  for (int i = tid; i < 16 * H_; i += 1024) {
    int r = i >> 8, j = i & 255;
    sA[r * KC_ + DIN_ + j] = (_Float16)0.0f;
    sC[i] = 0.0f;
  }
  __syncthreads();

  for (int t = 0; t < T_; ++t) {
    // stage x_t: 16 rows x 128 halves
    for (int i = tid; i < 16 * DIN_; i += 1024) {
      int r = i >> 7, k = i & 127;
      sA[r * KC_ + k] = seqF16[((size_t)(b0 + r) * T_ + t) * DIN_ + k];
    }
    if (t + 1 < T_ && tid < 16)  // global_prefetch next step's x
      __builtin_prefetch(&seqF16[((size_t)(b0 + tid) * T_ + t + 1) * DIN_], 0, 0);
    __syncthreads();

    // G[16 x 1024] = [x_t | h] @ Wcat^T  (12 K-steps of 32)
    if (cached) {
      #pragma unroll
      for (int tt = 0; tt < 2; ++tt) {
        const _Float16* wp = sW + (w * 32 + tt * 16) * KC_;
        v8f acc = {};
        #pragma unroll
        for (int kc = 0; kc < 12; ++kc) {
          v16h a = load_a_rm(sA + kc * 32, KC_, lane);
          v16h b = load_b_wt(wp + kc * 32, KC_, lane);
          acc = WMMA_F16(a, b, acc);
        }
        store_c(sG + (w * 2 + tt) * 16, G4_, lane, acc);
      }
    } else {
      #pragma unroll
      for (int tt = 0; tt < 2; ++tt) {
        const _Float16* wp = wcat + (size_t)(w * 32 + tt * 16) * KC_;
        v8f acc = {};
        #pragma unroll
        for (int kc = 0; kc < 12; ++kc) {
          v16h a = load_a_rm(sA + kc * 32, KC_, lane);
          v16h b = load_b_wt(wp + kc * 32, KC_, lane);
          acc = WMMA_F16(a, b, acc);
        }
        store_c(sG + (w * 2 + tt) * 16, G4_, lane, acc);
      }
    }
    __syncthreads();

    // gates (PyTorch order i,f,g,o) + state update
    for (int i = tid; i < 16 * H_; i += 1024) {
      int r = i >> 8, j = i & 255;
      float gi = sG[r * G4_ + j]       + bsum[j];
      float gf = sG[r * G4_ + 256 + j] + bsum[256 + j];
      float gg = sG[r * G4_ + 512 + j] + bsum[512 + j];
      float go = sG[r * G4_ + 768 + j] + bsum[768 + j];
      float c = sigf(gf) * sC[i] + sigf(gi) * tanhf(gg);
      float h = sigf(go) * tanhf(c);
      sC[i] = c;
      _Float16 hh = (_Float16)h;
      sA[r * KC_ + DIN_ + j] = hh;                               // next step A
      hsF16[((size_t)(b0 + r) * T_ + t) * H_ + j] = hh;          // outputs
      if (t == T_ - 1) hn[(b0 + r) * H_ + j] = h;
    }
    __syncthreads();
  }
}

// ---------------------------------------------------------------------------
// TPA: one WG (256 thr = 8 waves) per batch row.
// ---------------------------------------------------------------------------
__global__ void __launch_bounds__(256)
k_tpa(const _Float16* __restrict__ hsF16, const float* __restrict__ hn,
      const _Float16* __restrict__ wconv, const float* __restrict__ b_conv,
      const _Float16* __restrict__ wtpa,
      const float* __restrict__ W_ltd, const float* __restrict__ b_ltd,
      const float* __restrict__ W_tpa_h, const float* __restrict__ W_tpa_c,
      float* __restrict__ finOut, _Float16* __restrict__ finF16) {
  extern __shared__ char smem[];
  _Float16* sHS = (_Float16*)smem;                           // [128][256] 64KB
  _Float16* sHC = (_Float16*)(smem + T_ * H_ * 2);           // 8x[16][256] 64KB
  float* sHN = (float*)(smem + T_ * H_ * 2 + 8 * 16 * H_ * 2);
  float* sAl = sHN + H_;
  float* sVt = sAl + H_;

  const int b = blockIdx.x, tid = threadIdx.x, lane = tid & 31, w = tid >> 5;

  for (int i = tid; i < T_ * H_; i += 256)
    sHS[i] = hsF16[(size_t)b * T_ * H_ + i];
  for (int i = tid; i < H_; i += 256) {
    sHN[i] = hn[b * H_ + i]; sAl[i] = 0.0f; sVt[i] = 0.0f;
  }
  __syncthreads();

  for (int rep = 0; rep < 2; ++rep) {           // 16 i-chunks over 8 waves
    const int i0 = (w + rep * 8) * 16;
    _Float16* hc = sHC + w * 16 * H_;

    // hc chunk [16 x 256]: hc[m][j] = sum_t hs[t][i0+m]*Wconv[j][t] + b_conv[j]
    for (int jt = 0; jt < 16; ++jt) {
      v8f acc = {};
      #pragma unroll
      for (int kc = 0; kc < 4; ++kc) {
        v16h a;                                  // A[m][k] = hs[k][i0+m]
        #pragma unroll
        for (int h2 = 0; h2 < 16; ++h2)
          a[h2] = sHS[(kc * 32 + ak_idx(lane, h2)) * H_ + i0 + (lane & 15)];
        v16h bb = load_b_wt(wconv + (jt * 16) * T_ + kc * 32, T_, lane);
        acc = WMMA_F16(a, bb, acc);
      }
      int n = lane & 15, mb = (lane & 16) ? 8 : 0;
      #pragma unroll
      for (int v = 0; v < 8; ++v)
        hc[(mb + v) * H_ + jt * 16 + n] = (_Float16)(acc[v] + b_conv[jt * 16 + n]);
    }
    asm volatile("s_wait_dscnt 0" ::: "memory"); // hc visible wave-wide

    // alphaAcc[i0+m] += (hc @ Wtpa^T)[m][n] * hn[n]
    for (int nt = 0; nt < 16; ++nt) {
      v8f acc = {};
      #pragma unroll
      for (int kc = 0; kc < 8; ++kc) {
        v16h a = load_a_rm(hc + kc * 32, H_, lane);
        v16h bb = load_b_wt(wtpa + (nt * 16) * H_ + kc * 32, H_, lane);
        acc = WMMA_F16(a, bb, acc);
      }
      int n = nt * 16 + (lane & 15), mb = (lane & 16) ? 8 : 0;
      float hv = sHN[n];
      #pragma unroll
      for (int v = 0; v < 8; ++v)
        atomicAdd(&sAl[i0 + mb + v], acc[v] * hv);   // ds_add_f32
    }
    asm volatile("s_wait_dscnt 0" ::: "memory");

    // vt += sigmoid(alpha[i]) * hc[i][:]
    for (int m = 0; m < 16; ++m) {
      float al = sigf(sAl[i0 + m]);
      for (int j = lane; j < H_; j += 32)
        atomicAdd(&sVt[j], al * (float)hc[m * H_ + j]);
    }
    __syncthreads();
  }
  __syncthreads();

  // seq_inp = hn@W_ltd^T + b_ltd ; htprime = hn@W_tpa_h^T + vt@W_tpa_c^T
  if (tid < H_) {
    int n = tid;
    float s_ltd = b_ltd[n], s_h = 0.0f, s_c = 0.0f;
    for (int k = 0; k < H_; ++k) {
      float hv = sHN[k];
      s_ltd += W_ltd[n * H_ + k] * hv;
      s_h   += W_tpa_h[n * H_ + k] * hv;
      s_c   += W_tpa_c[n * H_ + k] * sVt[k];
    }
    float htp = s_h + s_c;
    size_t base = (size_t)b * FIN_;
    finOut[base + 160 + n] = s_ltd;
    finOut[base + 416 + n] = htp;
    finF16[base + 160 + n] = (_Float16)s_ltd;
    finF16[base + 416 + n] = (_Float16)htp;
  }
}

// ---------------------------------------------------------------------------
// MLP head: one wave per 16x16 output tile; C = relu(A@W^T + b), optional BN.
// ---------------------------------------------------------------------------
__global__ void __launch_bounds__(32)
k_head(const _Float16* __restrict__ A, const _Float16* __restrict__ W,
       const float* __restrict__ bias, const float* __restrict__ g,
       const float* __restrict__ be, const float* __restrict__ mu,
       const float* __restrict__ var, float* __restrict__ outF32,
       _Float16* __restrict__ outF16, int N, int K, int doBN) {
  const int lane = threadIdx.x & 31;
  const int m0 = blockIdx.x * 16, n0 = blockIdx.y * 16;
  v8f acc = {};
  for (int k0 = 0; k0 < K; k0 += 32) {
    v16h a = load_a_rm(A + (size_t)m0 * K + k0, K, lane);
    v16h b = load_b_wt(W + (size_t)n0 * K + k0, K, lane);
    acc = WMMA_F16(a, b, acc);
  }
  int n = n0 + (lane & 15), mb = (lane & 16) ? 8 : 0;
  float bn_s = 1.0f, bn_b = 0.0f;
  if (doBN) { bn_s = g[n] * rsqrtf(var[n] + 1e-5f); bn_b = be[n] - mu[n] * bn_s; }
  #pragma unroll
  for (int v = 0; v < 8; ++v) {
    int m = m0 + mb + v;
    float val = fmaxf(acc[v] + bias[n], 0.0f);
    if (doBN) val = val * bn_s + bn_b;
    if (outF32) outF32[(size_t)m * N + n] = val;
    if (outF16) outF16[(size_t)m * N + n] = (_Float16)val;
  }
}

// ---------------------------------------------------------------------------
extern "C" void kernel_launch(void* const* d_in, const int* in_sizes, int n_in,
                              void* d_out, int out_size, void* d_ws,
                              size_t ws_size, hipStream_t stream) {
  (void)in_sizes; (void)n_in; (void)out_size; (void)ws_size;
  const float* seq_cont = (const float*)d_in[0];
  const int*   seq_cat  = (const int*)d_in[1];
  const int*   ns_cat   = (const int*)d_in[2];
  const float* ns_cont  = (const float*)d_in[3];
  const float* emb_ns0  = (const float*)d_in[4];
  const float* emb_ns1  = (const float*)d_in[5];
  const float* emb_s0   = (const float*)d_in[6];
  const float* emb_s1   = (const float*)d_in[7];
  const float* W_ih     = (const float*)d_in[8];
  const float* W_hh     = (const float*)d_in[9];
  const float* b_ih     = (const float*)d_in[10];
  const float* b_hh     = (const float*)d_in[11];
  const float* W_ltd    = (const float*)d_in[12];
  const float* b_ltd    = (const float*)d_in[13];
  const float* W_conv   = (const float*)d_in[14];
  const float* b_conv   = (const float*)d_in[15];
  const float* W_tpa    = (const float*)d_in[16];
  const float* W_tpa_h  = (const float*)d_in[17];
  const float* W_tpa_c  = (const float*)d_in[18];
  const float* W_f1 = (const float*)d_in[19]; const float* b_f1 = (const float*)d_in[20];
  const float* g1 = (const float*)d_in[21]; const float* be1 = (const float*)d_in[22];
  const float* m1 = (const float*)d_in[23]; const float* v1 = (const float*)d_in[24];
  const float* W_f2 = (const float*)d_in[25]; const float* b_f2 = (const float*)d_in[26];
  const float* g2 = (const float*)d_in[27]; const float* be2 = (const float*)d_in[28];
  const float* m2 = (const float*)d_in[29]; const float* v2 = (const float*)d_in[30];
  const float* W_out = (const float*)d_in[31]; const float* b_out = (const float*)d_in[32];

  float* out_x   = (float*)d_out;                   // [B,64]
  float* out_fin = (float*)d_out + (size_t)B_ * 64; // [B,672]

  // ---- workspace carve (~107 MB) ----
  char* ws = (char*)d_ws; size_t o = 0;
  auto carve = [&](size_t bytes) {
    char* p = ws + o; o += (bytes + 255) & ~(size_t)255; return p;
  };
  _Float16* seqF16 = (_Float16*)carve((size_t)B_ * T_ * DIN_ * 2);
  _Float16* hsF16  = (_Float16*)carve((size_t)B_ * T_ * H_ * 2);
  float*    hn     = (float*)carve((size_t)B_ * H_ * 4);
  _Float16* wcat   = (_Float16*)carve((size_t)G4_ * KC_ * 2);
  float*    bsum   = (float*)carve(G4_ * 4);
  _Float16* wconv  = (_Float16*)carve((size_t)H_ * T_ * 2);
  _Float16* wtpa   = (_Float16*)carve((size_t)H_ * H_ * 2);
  _Float16* wf1    = (_Float16*)carve((size_t)512 * FIN_ * 2);
  _Float16* wf2    = (_Float16*)carve((size_t)256 * 512 * 2);
  _Float16* wout   = (_Float16*)carve((size_t)64 * 256 * 2);
  _Float16* finF16 = (_Float16*)carve((size_t)B_ * FIN_ * 2);
  _Float16* x1F16  = (_Float16*)carve((size_t)B_ * 512 * 2);
  _Float16* x2F16  = (_Float16*)carve((size_t)B_ * 256 * 2);

  // ---- setup ----
  hipLaunchKernelGGL(k_build_wcat, dim3(512), dim3(256), 0, stream,
                     W_ih, W_hh, b_ih, b_hh, wcat, bsum);
  hipLaunchKernelGGL(k_f32_to_f16, dim3(128), dim3(256), 0, stream, W_conv, wconv, H_ * T_);
  hipLaunchKernelGGL(k_f32_to_f16, dim3(256), dim3(256), 0, stream, W_tpa, wtpa, H_ * H_);
  hipLaunchKernelGGL(k_f32_to_f16, dim3(512), dim3(256), 0, stream, W_f1, wf1, 512 * FIN_);
  hipLaunchKernelGGL(k_f32_to_f16, dim3(512), dim3(256), 0, stream, W_f2, wf2, 256 * 512);
  hipLaunchKernelGGL(k_f32_to_f16, dim3(64), dim3(256), 0, stream, W_out, wout, 64 * 256);
  hipLaunchKernelGGL(k_embed_seq, dim3(4096), dim3(256), 0, stream,
                     seq_cont, seq_cat, emb_s0, emb_s1, seqF16);
  hipLaunchKernelGGL(k_embed_nonseq, dim3(640), dim3(256), 0, stream,
                     ns_cat, ns_cont, emb_ns0, emb_ns1, out_fin, finF16);

  // ---- LSTM recurrence ----
  size_t lstm_lds = (size_t)16 * KC_ * 2 + (size_t)16 * G4_ * 4 +
                    (size_t)16 * H_ * 4 + (size_t)CW_ * 32 * KC_ * 2; // 284KB
  hipLaunchKernelGGL(k_lstm, dim3(B_ / 16), dim3(1024), lstm_lds, stream,
                     seqF16, wcat, bsum, hsF16, hn);

  // ---- TPA attention (writes seq_inp + htprime into fin) ----
  size_t tpa_lds = (size_t)T_ * H_ * 2 + (size_t)8 * 16 * H_ * 2 + 3 * H_ * 4;
  hipLaunchKernelGGL(k_tpa, dim3(B_), dim3(256), tpa_lds, stream,
                     hsF16, hn, wconv, b_conv, wtpa, W_ltd, b_ltd,
                     W_tpa_h, W_tpa_c, out_fin, finF16);

  // ---- MLP head ----
  hipLaunchKernelGGL(k_head, dim3(B_ / 16, 512 / 16), dim3(32), 0, stream,
                     finF16, wf1, b_f1, g1, be1, m1, v1,
                     (float*)nullptr, x1F16, 512, FIN_, 1);
  hipLaunchKernelGGL(k_head, dim3(B_ / 16, 256 / 16), dim3(32), 0, stream,
                     x1F16, wf2, b_f2, g2, be2, m2, v2,
                     (float*)nullptr, x2F16, 256, 512, 1);
  hipLaunchKernelGGL(k_head, dim3(B_ / 16, 64 / 16), dim3(32), 0, stream,
                     x2F16, wout, b_out, (const float*)nullptr,
                     (const float*)nullptr, (const float*)nullptr,
                     (const float*)nullptr, out_x, (_Float16*)nullptr,
                     64, 256, 0);
}
